// QuantumSwarmCoordinator_12747462934642
// MI455X (gfx1250) — matrix operand order repaired and checked by reading, not compiled, not run on hardware
//
#include <hip/hip_runtime.h>
#include <hip/hip_bf16.h>
#include <math.h>

// ---------------------------------------------------------------------------
// QuantumSwarmCoordinator for MI455X (gfx1250, wave32, WMMA)
//   B=32, A=64, D=128, H=512, F=8192
// Core engine: v_wmma_f32_16x16x32_f16.
// Weights stored transposed f16 (BT[n][k]) so every WMMA operand fragment is
// exactly two 16-byte loads (layouts per CDNA5 ISA 7.12.2).
// Pair-MLP kernel stages h1 = relu(U+V+b1) in LDS once per 16-row tile and
// runs 4 WMMA accumulators per wave to amortize A-fragment traffic.
// ---------------------------------------------------------------------------

typedef __attribute__((ext_vector_type(16))) _Float16 v16h;
typedef __attribute__((ext_vector_type(8)))  float    v8f;
typedef __attribute__((ext_vector_type(4)))  float    f32x4;

union halfu {
    v16h     h;
    _Float16 e[16];
    f32x4    f[2];
};
union half8u {
    _Float16 e[8];
    f32x4    f;
};

#define QSC_B  32
#define QSC_A  64
#define QSC_D  128
#define QSC_H  512
#define QSC_F  8192

// ---------------------------------------------------------------------------
// Generic f16 WMMA GEMM:  C = act(A @ B + bias)
//   A  : row-major f16, M x K, row stride lda
//   BT : f16, N x K (B transposed), row stride ldb
//   C  : OutT, M x N  (row stride N)
//   ACT: 0 = none, 1 = relu, 2 = tanh
// One wave per 16x16 tile; block = 256 threads = 8 wave32.
// ---------------------------------------------------------------------------
template <int ACT, typename OutT>
__global__ void qsc_gemm_f16(const _Float16* __restrict__ A,
                             const _Float16* __restrict__ BT,
                             const float* __restrict__ bias,
                             OutT* __restrict__ C,
                             int N, int K, int lda, int ldb) {
    const int wave   = threadIdx.x >> 5;
    const int lane   = threadIdx.x & 31;
    const int laneLo = lane & 15;
    const int laneHi = lane >> 4;
    const int tm = blockIdx.x;
    const int tn = blockIdx.y * 8 + wave;
    if (tn * 16 >= N) return;  // wave-uniform: EXEC stays all-ones for WMMA

    const _Float16* Ar = A  + (size_t)(tm * 16 + laneLo) * lda;
    const _Float16* Br = BT + (size_t)(tn * 16 + laneLo) * ldb;

    v8f acc = {};
    for (int k0 = 0; k0 < K; k0 += 32) {
        halfu au, bu;
        // A frag: halves 0..7 = K[k0+8*laneHi .. +7], halves 8..15 = +16
        const _Float16* ap = Ar + k0 + laneHi * 8;
        au.f[0] = *(const f32x4*)(ap);
        au.f[1] = *(const f32x4*)(ap + 16);
        // B frag: halves 0..7 = rows k0+8*laneHi .. +7, halves 8..15 = +16, at column n
        const _Float16* bp = Br + k0 + laneHi * 8;
        bu.f[0] = *(const f32x4*)(bp);
        bu.f[1] = *(const f32x4*)(bp + 16);
        acc = __builtin_amdgcn_wmma_f32_16x16x32_f16(
            false, au.h, false, bu.h, (short)0, acc, false, false);
    }

    const float bn = bias ? bias[tn * 16 + laneLo] : 0.0f;
#pragma unroll
    for (int r = 0; r < 8; ++r) {
        float v = acc[r] + bn;
        if (ACT == 1) v = fmaxf(v, 0.0f);
        if (ACT == 2) v = tanhf(v);
        const int mm = tm * 16 + laneHi * 8 + r;
        C[(size_t)mm * N + tn * 16 + laneLo] = (OutT)v;
    }
}

// ---------------------------------------------------------------------------
// Fused pairwise-MLP GEMM:
//   row r = (b*A + i)*A + j,  h1[r,k] = relu(U[b*A+i,k] + V[b*A+j,k] + b1[k])
//   H2 = relu(h1 @ Wi2 + b2)  -> f16, M=131072, K=N=512
// One block per 16-row tile. Phase 1: 256 threads build the 16x512 h1 tile
// in LDS (f16, padded stride). Phase 2: 8 waves x 4 N-tiles, A-fragments
// from LDS (2 x ds_load_b128), 4 WMMAs per A-fragment.
// ---------------------------------------------------------------------------
#define H1_STRIDE 520   // 512 + 8 halves: rows 1040B apart -> conflict-free b128

__global__ void qsc_pair_gemm(const float* __restrict__ U,
                              const float* __restrict__ V,
                              const float* __restrict__ b1,
                              const _Float16* __restrict__ W2T,
                              const float* __restrict__ b2,
                              _Float16* __restrict__ H2) {
    __shared__ _Float16 h1s[16 * H1_STRIDE];

    const int Kd = QSC_H, Nd = QSC_H;
    const int tm = blockIdx.x;

    // ---- Phase 1: h1 tile -> LDS (each thread: one row, 32 k-values) ----
    {
        const int mRow  = threadIdx.x >> 4;        // 0..15
        const int kslot = (threadIdx.x & 15) * 32; // 0,32,..,480
        const int r  = tm * 16 + mRow;
        const int bi = r >> 6;                     // b*A + i
        const int j  = r & 63;
        const int bj = (r >> 12) * 64 + j;         // b*A + j
        const float* Ur = U  + (size_t)bi * QSC_H + kslot;
        const float* Vr = V  + (size_t)bj * QSC_H + kslot;
        const float* br = b1 + kslot;
        _Float16* dst = &h1s[mRow * H1_STRIDE + kslot];
#pragma unroll
        for (int c = 0; c < 32; c += 8) {
            const f32x4 u0 = *(const f32x4*)(Ur + c);
            const f32x4 u1 = *(const f32x4*)(Ur + c + 4);
            const f32x4 v0 = *(const f32x4*)(Vr + c);
            const f32x4 v1 = *(const f32x4*)(Vr + c + 4);
            const f32x4 c0 = *(const f32x4*)(br + c);
            const f32x4 c1 = *(const f32x4*)(br + c + 4);
            half8u o;
#pragma unroll
            for (int t = 0; t < 4; ++t) {
                o.e[t]     = (_Float16)fmaxf(u0[t] + v0[t] + c0[t], 0.0f);
                o.e[4 + t] = (_Float16)fmaxf(u1[t] + v1[t] + c1[t], 0.0f);
            }
            *(f32x4*)(dst + c) = o.f;
        }
    }
    __syncthreads();

    // ---- Phase 2: 8 waves x 4 N-tiles, WMMA from LDS/global ----
    const int wave   = threadIdx.x >> 5;
    const int lane   = threadIdx.x & 31;
    const int laneLo = lane & 15;
    const int laneHi = lane >> 4;

    const _Float16* aBase = &h1s[laneLo * H1_STRIDE + laneHi * 8];
    const _Float16* bBase[4];
#pragma unroll
    for (int t = 0; t < 4; ++t) {
        const int tn = wave + t * 8;
        bBase[t] = W2T + (size_t)(tn * 16 + laneLo) * Kd + laneHi * 8;
    }

    v8f acc0 = {}, acc1 = {}, acc2 = {}, acc3 = {};
    for (int k0 = 0; k0 < Kd; k0 += 32) {
        halfu au;
        au.f[0] = *(const f32x4*)(aBase + k0);
        au.f[1] = *(const f32x4*)(aBase + k0 + 16);
        halfu b0, b1f, b2f, b3f;
        b0.f[0]  = *(const f32x4*)(bBase[0] + k0);
        b0.f[1]  = *(const f32x4*)(bBase[0] + k0 + 16);
        b1f.f[0] = *(const f32x4*)(bBase[1] + k0);
        b1f.f[1] = *(const f32x4*)(bBase[1] + k0 + 16);
        b2f.f[0] = *(const f32x4*)(bBase[2] + k0);
        b2f.f[1] = *(const f32x4*)(bBase[2] + k0 + 16);
        b3f.f[0] = *(const f32x4*)(bBase[3] + k0);
        b3f.f[1] = *(const f32x4*)(bBase[3] + k0 + 16);
        acc0 = __builtin_amdgcn_wmma_f32_16x16x32_f16(false, au.h, false, b0.h,  (short)0, acc0, false, false);
        acc1 = __builtin_amdgcn_wmma_f32_16x16x32_f16(false, au.h, false, b1f.h, (short)0, acc1, false, false);
        acc2 = __builtin_amdgcn_wmma_f32_16x16x32_f16(false, au.h, false, b2f.h, (short)0, acc2, false, false);
        acc3 = __builtin_amdgcn_wmma_f32_16x16x32_f16(false, au.h, false, b3f.h, (short)0, acc3, false, false);
    }

    // ---- Epilogue ----
    v8f* accs[4] = { &acc0, &acc1, &acc2, &acc3 };
#pragma unroll
    for (int t = 0; t < 4; ++t) {
        const int tn = wave + t * 8;
        const float bn = b2[tn * 16 + laneLo];
        const v8f a = *accs[t];
#pragma unroll
        for (int rr = 0; rr < 8; ++rr) {
            const float v = fmaxf(a[rr] + bn, 0.0f);
            const int mm = tm * 16 + laneHi * 8 + rr;
            H2[(size_t)mm * Nd + tn * 16 + laneLo] = (_Float16)v;
        }
    }
}

// ---------------------------------------------------------------------------
// S[bi,k] = sum_{j != i} H2[bi*A + j, k]   (f16 in, f32 accumulate, f16 out)
// ---------------------------------------------------------------------------
__global__ void qsc_reduce_h2(const _Float16* __restrict__ H2,
                              _Float16* __restrict__ S) {
    const int idx = blockIdx.x * blockDim.x + threadIdx.x;  // 2048*512 threads
    const int k  = idx & (QSC_H - 1);
    const int bi = idx >> 9;
    const int i  = bi & 63;
    const _Float16* base = H2 + ((size_t)bi * 64) * QSC_H + k;
    float sum = 0.0f;
#pragma unroll 8
    for (int j = 0; j < QSC_A; ++j) {
        const float v = (float)base[(size_t)j * QSC_H];
        sum += (j == i) ? 0.0f : v;
    }
    S[idx] = (_Float16)sum;
}

// ---------------------------------------------------------------------------
// Transpose + f32->f16 convert:  BT[n*K + k] = (f16) W[k*N + n]
// block (32,8), 32x32 LDS tile; K, N multiples of 32 here.
// ---------------------------------------------------------------------------
__global__ void qsc_transpose_cvt(const float* __restrict__ W,
                                  _Float16* __restrict__ BT,
                                  int K, int N) {
    __shared__ _Float16 tile[32][33];
    const int kBase = blockIdx.x * 32;
    const int nBase = blockIdx.y * 32;
#pragma unroll
    for (int r = threadIdx.y; r < 32; r += 8) {
        tile[r][threadIdx.x] =
            (_Float16)W[(size_t)(kBase + r) * N + nBase + threadIdx.x];
    }
    __syncthreads();
#pragma unroll
    for (int r = threadIdx.y; r < 32; r += 8) {
        BT[(size_t)(nBase + r) * K + kBase + threadIdx.x] = tile[threadIdx.x][r];
    }
}

__global__ void qsc_cvt_f16(const float* __restrict__ in,
                            _Float16* __restrict__ out, int n) {
    const int i = blockIdx.x * blockDim.x + threadIdx.x;
    if (i < n) out[i] = (_Float16)in[i];
}

// ---------------------------------------------------------------------------
// Final combine:
//   interactions = interSum/63 + bi3
//   coordinated  = states + 0.1*interactions + 0.1*coordination
// ---------------------------------------------------------------------------
__global__ void qsc_combine(const float* __restrict__ states,
                            const float* __restrict__ interSum,
                            const float* __restrict__ bi3,
                            const float* __restrict__ coord,
                            float* __restrict__ outCoordinated,
                            float* __restrict__ outInteractions) {
    const int idx = blockIdx.x * blockDim.x + threadIdx.x;  // 262144 threads
    const int d = idx & (QSC_D - 1);
    const float inter = interSum[idx] * (1.0f / 63.0f) + bi3[d];
    outInteractions[idx] = inter;
    outCoordinated[idx]  = states[idx] + 0.1f * inter + 0.1f * coord[idx];
}

// ---------------------------------------------------------------------------
// Host launcher
// ---------------------------------------------------------------------------
extern "C" void kernel_launch(void* const* d_in, const int* in_sizes, int n_in,
                              void* d_out, int out_size, void* d_ws, size_t ws_size,
                              hipStream_t stream) {
    const float* states = (const float*)d_in[0];   // (B,A,D)
    // d_in[1] = swarm_goals (unused by reference)
    const float* Wq  = (const float*)d_in[2];
    const float* bq  = (const float*)d_in[3];
    const float* Wi1 = (const float*)d_in[4];
    const float* bi1 = (const float*)d_in[5];
    const float* Wi2 = (const float*)d_in[6];
    const float* bi2 = (const float*)d_in[7];
    const float* Wi3 = (const float*)d_in[8];
    const float* bi3 = (const float*)d_in[9];
    const float* Wc1 = (const float*)d_in[10];
    const float* bc1 = (const float*)d_in[11];
    const float* Wc2 = (const float*)d_in[12];
    const float* bc2 = (const float*)d_in[13];
    const float* Wc3 = (const float*)d_in[14];
    const float* bc3 = (const float*)d_in[15];

    float* out = (float*)d_out;
    float* outCoordinated  = out;                 // (B,A,D)   262144
    float* outQuantum      = out + 262144;        // (B,F)     262144
    float* outInteractions = out + 524288;        // (B,A,D)   262144
    float* outCoordination = out + 786432;        // (B,A,D)   262144

    // Workspace carve-up (all sizes are 256B multiples)
    char* ws = (char*)d_ws;
    size_t off = 0;
    auto carve = [&](size_t bytes) { char* p = ws + off; off += bytes; return p; };
    _Float16* WqT   = (_Float16*)carve((size_t)QSC_F * QSC_F * 2);   // 8192x8192
    _Float16* Wc1T  = (_Float16*)carve((size_t)QSC_H * QSC_F * 2);   // 512x8192
    _Float16* Wc3T  = (_Float16*)carve((size_t)QSC_F * QSC_H * 2);   // 8192x512
    _Float16* Wi1T  = (_Float16*)carve((size_t)QSC_H * 256 * 2);     // 512x256
    _Float16* Wi2T  = (_Float16*)carve((size_t)QSC_H * QSC_H * 2);
    _Float16* Wi3T  = (_Float16*)carve((size_t)QSC_D * QSC_H * 2);   // 128x512
    _Float16* Wc2T  = (_Float16*)carve((size_t)QSC_H * QSC_H * 2);
    _Float16* stF16 = (_Float16*)carve((size_t)262144 * 2);
    _Float16* QF16  = (_Float16*)carve((size_t)262144 * 2);
    float*    U     = (float*)   carve((size_t)2048 * QSC_H * 4);
    float*    V     = (float*)   carve((size_t)2048 * QSC_H * 4);
    _Float16* H2    = (_Float16*)carve((size_t)131072 * QSC_H * 2);
    _Float16* Sred  = (_Float16*)carve((size_t)2048 * QSC_H * 2);
    float*    iSum  = (float*)   carve((size_t)2048 * QSC_D * 4);
    _Float16* c1    = (_Float16*)carve((size_t)32 * QSC_H * 2);
    _Float16* c2    = (_Float16*)carve((size_t)32 * QSC_H * 2);
    (void)ws_size; (void)n_in; (void)in_sizes; (void)out_size;

    const dim3 tb(32, 8);
    // --- weight conversions (transpose + f16) -----------------------------
    qsc_transpose_cvt<<<dim3(256, 256), tb, 0, stream>>>(Wq,  WqT,  QSC_F, QSC_F);
    qsc_transpose_cvt<<<dim3(8,   16),  tb, 0, stream>>>(Wi1, Wi1T, 256,   QSC_H);
    qsc_transpose_cvt<<<dim3(16,  16),  tb, 0, stream>>>(Wi2, Wi2T, QSC_H, QSC_H);
    qsc_transpose_cvt<<<dim3(16,  4),   tb, 0, stream>>>(Wi3, Wi3T, QSC_H, QSC_D);
    qsc_transpose_cvt<<<dim3(256, 16),  tb, 0, stream>>>(Wc1, Wc1T, QSC_F, QSC_H);
    qsc_transpose_cvt<<<dim3(16,  16),  tb, 0, stream>>>(Wc2, Wc2T, QSC_H, QSC_H);
    qsc_transpose_cvt<<<dim3(16,  256), tb, 0, stream>>>(Wc3, Wc3T, QSC_H, QSC_F);
    qsc_cvt_f16<<<1024, 256, 0, stream>>>(states, stF16, 262144);

    // --- quantum features: Q = tanh(states_flat @ Wq + bq)  (32 x 8192) ---
    qsc_gemm_f16<2, float><<<dim3(2, 64), 256, 0, stream>>>(
        stF16, WqT, bq, outQuantum, QSC_F, QSC_F, QSC_F, QSC_F);
    qsc_cvt_f16<<<1024, 256, 0, stream>>>(outQuantum, QF16, 262144);

    // --- pairwise layer 1 factorization: U = states@Wi1[:D], V = @Wi1[D:] -
    qsc_gemm_f16<0, float><<<dim3(128, 4), 256, 0, stream>>>(
        stF16, Wi1T,       nullptr, U, QSC_H, QSC_D, QSC_D, 256);
    qsc_gemm_f16<0, float><<<dim3(128, 4), 256, 0, stream>>>(
        stF16, Wi1T + 128, nullptr, V, QSC_H, QSC_D, QSC_D, 256);

    // --- fused pair GEMM: H2 = relu(relu(U+V+b1) @ Wi2 + b2) --------------
    qsc_pair_gemm<<<dim3(8192), 256, 0, stream>>>(U, V, bi1, Wi2T, bi2, H2);

    // --- reduce over j != i, then linear layer 3 --------------------------
    qsc_reduce_h2<<<4096, 256, 0, stream>>>(H2, Sred);
    qsc_gemm_f16<0, float><<<dim3(128, 1), 256, 0, stream>>>(
        Sred, Wi3T, nullptr, iSum, QSC_D, QSC_H, QSC_H, QSC_H);

    // --- coordination net --------------------------------------------------
    qsc_gemm_f16<1, _Float16><<<dim3(2, 4), 256, 0, stream>>>(
        QF16, Wc1T, bc1, c1, QSC_H, QSC_F, QSC_F, QSC_F);
    qsc_gemm_f16<1, _Float16><<<dim3(2, 4), 256, 0, stream>>>(
        c1, Wc2T, bc2, c2, QSC_H, QSC_H, QSC_H, QSC_H);
    qsc_gemm_f16<0, float><<<dim3(2, 64), 256, 0, stream>>>(
        c2, Wc3T, bc3, outCoordination, QSC_F, QSC_H, QSC_H, QSC_H);

    // --- final combine ------------------------------------------------------
    qsc_combine<<<1024, 256, 0, stream>>>(
        states, iSum, bi3, outCoordination, outCoordinated, outInteractions);
}
